// Attention_6914897346741
// MI455X (gfx1250) — matrix operand order repaired and checked by reading, not compiled, not run on hardware
//
#include <hip/hip_runtime.h>
#include <hip/hip_bf16.h>

// ---------------------------------------------------------------------------
// GQA attention layer for MI455X (gfx1250), wave32 + WMMA f16 (f32 accum).
//   x:[B,L,D] f32, Wq:[D,H*HD], Wk/Wv:[D,KV*HD], Wo:[H*HD,D] -> out:[B,L,D] f32
// Internals in f16 (WMMA 16x16x32), accumulation in f32.
// ---------------------------------------------------------------------------

typedef __attribute__((ext_vector_type(16))) _Float16 v16h;
typedef __attribute__((ext_vector_type(8)))  _Float16 v8h;
typedef __attribute__((ext_vector_type(8)))  float    v8f;

static constexpr int Bc    = 2;
static constexpr int Lc    = 2048;
static constexpr int Dc    = 3072;
static constexpr int Hc    = 24;
static constexpr int KVc   = 8;
static constexpr int HDc   = 128;
static constexpr int GRPc  = 3;          // H / KV
static constexpr int LPc   = Lc + 16;    // padded key length for Vt columns
static constexpr float THETAc = 500000.0f;
static constexpr float SCALEc = 0.08838834764831845f; // 128^-0.5

static __device__ __forceinline__ v16h cat16(v8h lo, v8h hi) {
  return __builtin_shufflevector(lo, hi, 0,1,2,3,4,5,6,7,8,9,10,11,12,13,14,15);
}
static __device__ __forceinline__ v8f zero8() {
  v8f z;
#pragma unroll
  for (int i = 0; i < 8; ++i) z[i] = 0.0f;
  return z;
}
static __device__ __forceinline__ v8f wmma16(v16h a, v16h b, v8f c) {
  return __builtin_amdgcn_wmma_f32_16x16x32_f16(false, a, false, b, (short)0, c,
                                                false, false);
}

// --------------------------- elementwise helpers ---------------------------

__global__ __launch_bounds__(256) void cast_f32_f16_kernel(
    const float* __restrict__ in, _Float16* __restrict__ out, size_t n) {
  size_t i = (size_t)blockIdx.x * blockDim.x + threadIdx.x;
  if (i < n) out[i] = (_Float16)in[i];
}

// in: [K, N] f32 (row-major) -> out: [N, K] f16 (row-major) == B transposed
__global__ __launch_bounds__(256) void transpose_w_kernel(
    const float* __restrict__ in, _Float16* __restrict__ out, int K, int N) {
  size_t i = (size_t)blockIdx.x * blockDim.x + threadIdx.x;
  if (i >= (size_t)K * N) return;
  int k = (int)(i / N);
  int n = (int)(i % N);
  out[(size_t)n * K + k] = (_Float16)in[i];
}

// Vf32: [B, L, KV, HD] -> Vt f16: [B, KV, HD, LP]
__global__ __launch_bounds__(256) void transpose_v_kernel(
    const float* __restrict__ in, _Float16* __restrict__ out) {
  size_t i = (size_t)blockIdx.x * blockDim.x + threadIdx.x;
  if (i >= (size_t)Bc * Lc * KVc * HDc) return;
  int hd = (int)(i % HDc);
  size_t t = i / HDc;
  int kv = (int)(t % KVc); t /= KVc;
  int l  = (int)(t % Lc);
  int b  = (int)(t / Lc);
  out[(((size_t)b * KVc + kv) * HDc + hd) * LPc + l] = (_Float16)in[i];
}

// Half-split llama RoPE, f32 -> f16.  X: [B, L, NH, HD]
__global__ __launch_bounds__(256) void rope_f16_kernel(
    const float* __restrict__ X, _Float16* __restrict__ Y, int NH) {
  size_t i = (size_t)blockIdx.x * blockDim.x + threadIdx.x;
  size_t total = (size_t)Bc * Lc * NH * (HDc / 2);
  if (i >= total) return;
  int hd = (int)(i % (HDc / 2));
  size_t rest = i / (HDc / 2);
  int h = (int)(rest % NH);
  size_t bl = rest / NH;          // b*L + l
  int l = (int)(bl % Lc);
  float inv = __powf(THETAc, -(float)hd / (float)(HDc / 2));
  float ang = (float)l * inv;
  float c = __cosf(ang), s = __sinf(ang);
  size_t base = (bl * NH + h) * (size_t)HDc;
  float x1 = X[base + hd];
  float x2 = X[base + HDc / 2 + hd];
  Y[base + hd]           = (_Float16)(x1 * c - x2 * s);
  Y[base + HDc / 2 + hd] = (_Float16)(x2 * c + x1 * s);
}

// ------------------------------- WMMA GEMM ---------------------------------
// C[M,N] f32 = A[M,K] f16 (row-major) x Bt[N,K] f16 (B transposed, row-major)
// One wave computes a 32x64 tile (2 A-fragments x 4 B-fragments = 8 WMMA per
// K-step).  Software pipelined: next K-step's 6 fragments are loaded into
// distinct registers before the current WMMAs issue, so the per-wave waits
// become partial (loadcnt<=N) instead of full barriers.
__global__ __launch_bounds__(256) void wmma_gemm_kernel(
    const _Float16* __restrict__ A, const _Float16* __restrict__ Bt,
    float* __restrict__ C, int M, int N, int K) {
  const int lane = threadIdx.x & 31;
  const int wave = threadIdx.x >> 5;
  const int ntiles = N >> 6;
  const int task = blockIdx.x * 8 + wave;
  if (task >= (M >> 5) * ntiles) return;   // wave-uniform exit (EXEC stays full)
  const int tm = task / ntiles;
  const int tn = task % ntiles;
  const int mrow = lane & 15;   // A fragment: lane holds matrix row M
  const int ph   = lane >> 4;   // half-wave selector

  // A frag element e -> k = k0 + ph*16 + e: one contiguous 32B run
  const _Float16* arow0 = A + (size_t)(tm * 32 + mrow) * K + ph * 16;
  const _Float16* arow1 = arow0 + (size_t)16 * K;
  // B frag (col = lane&15): k = k0 + ph*16 + e: one contiguous 32B run
  const _Float16* brow[4];
#pragma unroll
  for (int t = 0; t < 4; ++t)
    brow[t] = Bt + (size_t)(tn * 64 + t * 16 + mrow) * K + ph * 16;

  v8f acc[2][4];
#pragma unroll
  for (int r = 0; r < 2; ++r)
#pragma unroll
    for (int t = 0; t < 4; ++t) acc[r][t] = zero8();

  auto ldA = [&](const _Float16* p, int k0) {
    return cat16(*(const v8h*)(p + k0), *(const v8h*)(p + k0 + 8));
  };
  auto ldB = [&](int t, int k0) {
    return cat16(*(const v8h*)(brow[t] + k0), *(const v8h*)(brow[t] + k0 + 8));
  };

  v16h a0 = ldA(arow0, 0), a1 = ldA(arow1, 0);
  v16h b0 = ldB(0, 0), b1 = ldB(1, 0), b2 = ldB(2, 0), b3 = ldB(3, 0);

  for (int k0 = 0; k0 < K - 32; k0 += 32) {
    const int kn = k0 + 32;
    __builtin_prefetch(arow0 + k0 + 2048, 0, 3);   // global_prefetch_b8
    // issue next-step loads first (distinct regs -> stay in flight)
    v16h a0n = ldA(arow0, kn), a1n = ldA(arow1, kn);
    v16h b0n = ldB(0, kn), b1n = ldB(1, kn), b2n = ldB(2, kn), b3n = ldB(3, kn);
    // 8 WMMAs on the previous step's fragments
    acc[0][0] = wmma16(a0, b0, acc[0][0]);
    acc[1][0] = wmma16(a1, b0, acc[1][0]);
    acc[0][1] = wmma16(a0, b1, acc[0][1]);
    acc[1][1] = wmma16(a1, b1, acc[1][1]);
    acc[0][2] = wmma16(a0, b2, acc[0][2]);
    acc[1][2] = wmma16(a1, b2, acc[1][2]);
    acc[0][3] = wmma16(a0, b3, acc[0][3]);
    acc[1][3] = wmma16(a1, b3, acc[1][3]);
    a0 = a0n; a1 = a1n; b0 = b0n; b1 = b1n; b2 = b2n; b3 = b3n;
  }
  // drain: last K-step
  acc[0][0] = wmma16(a0, b0, acc[0][0]);
  acc[1][0] = wmma16(a1, b0, acc[1][0]);
  acc[0][1] = wmma16(a0, b1, acc[0][1]);
  acc[1][1] = wmma16(a1, b1, acc[1][1]);
  acc[0][2] = wmma16(a0, b2, acc[0][2]);
  acc[1][2] = wmma16(a1, b2, acc[1][2]);
  acc[0][3] = wmma16(a0, b3, acc[0][3]);
  acc[1][3] = wmma16(a1, b3, acc[1][3]);

  // C/D layout: VGPR v holds row (v + ph*8), lane&15 is column
#pragma unroll
  for (int r = 0; r < 2; ++r)
#pragma unroll
    for (int t = 0; t < 4; ++t)
#pragma unroll
      for (int v = 0; v < 8; ++v)
        C[(size_t)(tm * 32 + r * 16 + v + ph * 8) * N + tn * 64 + t * 16 + mrow] =
            acc[r][t][v];
}

// ----------------------------- flash attention -----------------------------
// Qh: [B, L, H, HD] f16 (roped) ; Kh: [B, L, KV, HD] f16 (roped)
// Vt: [B, KV, HD, LP] f16       ; Oh: [B, L, H, HD] f16
// One wave owns one (head, 16-query tile); iterates keys 32 at a time with
// online softmax.  8 QK^T WMMAs + 8 PV WMMAs per key step; fragments are
// preloaded in bulk so the two accumulator chains overlap the loads.
__global__ __launch_bounds__(256) void flash_attn_kernel(
    const _Float16* __restrict__ Qh, const _Float16* __restrict__ Kh,
    const _Float16* __restrict__ Vt, _Float16* __restrict__ Oh) {
  __shared__ __align__(16) _Float16 pbuf[8][16][32];   // per-wave P staging
  const int lane = threadIdx.x & 31;
  const int wave = threadIdx.x >> 5;
  const int QT = Lc / 16;
  const int task = blockIdx.x * 8 + wave;
  if (task >= Bc * Hc * QT) return;
  const int qt   = task % QT;
  const int head = task / QT;
  const int b  = head / Hc;
  const int h  = head % Hc;
  const int kv = h / GRPc;
  const int cl = lane & 15;
  const int ph = lane >> 4;

  // Q A-fragments: 16x128 tile -> 4 chunks of K=32
  const _Float16* qp = Qh + (((size_t)b * Lc + qt * 16 + cl) * Hc + h) * HDc + ph * 16;
  v16h qf[4];
#pragma unroll
  for (int c = 0; c < 4; ++c)
    qf[c] = cat16(*(const v8h*)(qp + c * 32), *(const v8h*)(qp + c * 32 + 8));

  const _Float16* kbase = Kh + ((size_t)b * Lc * KVc + kv) * HDc + ph * 16;
  const _Float16* vbase = Vt + (((size_t)b * KVc + kv) * HDc + cl) * LPc + ph * 16;

  v8f accO[8];
#pragma unroll
  for (int t = 0; t < 8; ++t) accO[t] = zero8();
  float m_run[8], l_run[8];
#pragma unroll
  for (int v = 0; v < 8; ++v) { m_run[v] = -3.0e38f; l_run[v] = 0.0f; }

  const int kend = qt * 16 + 16;           // causal: keys [0, kend)
  for (int kt = 0; kt < kend; kt += 32) {
    const int k0c = kt + cl;               // key for score subtile 0
    const int k1c = kt + 16 + cl;          // key for score subtile 1
    const int kk0 = k0c < Lc ? k0c : Lc - 1;
    const int kk1 = k1c < Lc ? k1c : Lc - 1;
    const _Float16* kp0 = kbase + (size_t)kk0 * (KVc * HDc);
    const _Float16* kp1 = kbase + (size_t)kk1 * (KVc * HDc);
    // Preload all 8 K fragments (distinct regs -> overlapping loads)
    v16h kf0[4], kf1[4];
#pragma unroll
    for (int c = 0; c < 4; ++c) {
      kf0[c] = cat16(*(const v8h*)(kp0 + c * 32), *(const v8h*)(kp0 + c * 32 + 8));
      kf1[c] = cat16(*(const v8h*)(kp1 + c * 32), *(const v8h*)(kp1 + c * 32 + 8));
    }
    v8f s0 = zero8(), s1 = zero8();
#pragma unroll
    for (int c = 0; c < 4; ++c) {          // two independent accumulator chains
      s0 = wmma16(qf[c], kf0[c], s0);
      s1 = wmma16(qf[c], kf1[c], s1);
    }
    // Online softmax over the 16x32 score tile (rows live across 16-lane halves)
#pragma unroll
    for (int v = 0; v < 8; ++v) {
      const int r = qt * 16 + v + ph * 8;
      float a0 = s0[v] * SCALEc; if (k0c > r) a0 = -3.0e38f;
      float a1 = s1[v] * SCALEc; if (k1c > r) a1 = -3.0e38f;
      float mloc = fmaxf(a0, a1);
      mloc = fmaxf(mloc, __shfl_xor(mloc, 1));
      mloc = fmaxf(mloc, __shfl_xor(mloc, 2));
      mloc = fmaxf(mloc, __shfl_xor(mloc, 4));
      mloc = fmaxf(mloc, __shfl_xor(mloc, 8));
      const float mn = fmaxf(m_run[v], mloc);
      const float p0 = __expf(a0 - mn);
      const float p1 = __expf(a1 - mn);
      float ls = p0 + p1;
      ls += __shfl_xor(ls, 1); ls += __shfl_xor(ls, 2);
      ls += __shfl_xor(ls, 4); ls += __shfl_xor(ls, 8);
      const float alpha = __expf(m_run[v] - mn);
      l_run[v] = l_run[v] * alpha + ls;
      m_run[v] = mn;
#pragma unroll
      for (int t = 0; t < 8; ++t) accO[t][v] *= alpha;
      pbuf[wave][v + ph * 8][cl]      = (_Float16)p0;
      pbuf[wave][v + ph * 8][16 + cl] = (_Float16)p1;
    }
    // Wave-local LDS RAW: DS ops are in-order per wave; wait then reread in
    // A-fragment layout (row = lane&15, k = ph*16 + {0..15}).
    asm volatile("s_wait_dscnt 0x0" ::: "memory");
    const _Float16* pp = &pbuf[wave][cl][ph * 16];
    v16h pf = cat16(*(const v8h*)pp, *(const v8h*)(pp + 8));
    // PV: process hd-tiles in two groups of 4; preload each group's V frags
#pragma unroll
    for (int g = 0; g < 2; ++g) {
      v16h vf[4];
#pragma unroll
      for (int u = 0; u < 4; ++u) {
        const _Float16* vp = vbase + (size_t)((g * 4 + u) * 16) * LPc + kt;
        vf[u] = cat16(*(const v8h*)vp, *(const v8h*)(vp + 8));
      }
#pragma unroll
      for (int u = 0; u < 4; ++u)          // 4 independent accumulator chains
        accO[g * 4 + u] = wmma16(pf, vf[u], accO[g * 4 + u]);
    }
  }
  // Normalize and store f16 output tile [B, L, H, HD]
#pragma unroll
  for (int t = 0; t < 8; ++t)
#pragma unroll
    for (int v = 0; v < 8; ++v) {
      const int qrow = qt * 16 + v + ph * 8;
      Oh[(((size_t)b * Lc + qrow) * Hc + h) * HDc + t * 16 + cl] =
          (_Float16)(accO[t][v] / l_run[v]);
    }
}

// ------------------------------- launcher ----------------------------------

extern "C" void kernel_launch(void* const* d_in, const int* in_sizes, int n_in,
                              void* d_out, int out_size, void* d_ws, size_t ws_size,
                              hipStream_t stream) {
  (void)in_sizes; (void)n_in; (void)out_size; (void)ws_size;
  const float* x  = (const float*)d_in[0];
  const float* Wq = (const float*)d_in[1];
  const float* Wk = (const float*)d_in[2];
  const float* Wv = (const float*)d_in[3];
  const float* Wo = (const float*)d_in[4];
  float* out = (float*)d_out;

  const size_t ML = (size_t)Bc * Lc;          // 4096 rows
  char* p = (char*)d_ws;
  auto carve = [&](size_t bytes) -> char* {
    char* q = p; p += (bytes + 255) & ~(size_t)255; return q;
  };
  _Float16* xh   = (_Float16*)carve(ML * Dc * 2);                       // x f16
  _Float16* wqt  = (_Float16*)carve((size_t)(Hc * HDc) * Dc * 2);       // Wq^T
  _Float16* wkt  = (_Float16*)carve((size_t)(KVc * HDc) * Dc * 2);      // Wk^T
  _Float16* wvt  = (_Float16*)carve((size_t)(KVc * HDc) * Dc * 2);      // Wv^T
  _Float16* wot  = (_Float16*)carve((size_t)Dc * (Hc * HDc) * 2);       // Wo^T
  float*    qf32 = (float*)carve(ML * Hc * HDc * 4);
  float*    kf32 = (float*)carve(ML * KVc * HDc * 4);
  float*    vf32 = (float*)carve(ML * KVc * HDc * 4);
  _Float16* qh   = (_Float16*)carve(ML * Hc * HDc * 2);
  _Float16* kh   = (_Float16*)carve(ML * KVc * HDc * 2);
  _Float16* vt   = (_Float16*)carve((size_t)Bc * KVc * HDc * LPc * 2);
  _Float16* ah   = (_Float16*)carve(ML * Hc * HDc * 2);                 // attn out

  auto blocks = [](size_t n) { return (unsigned)((n + 255) / 256); };

  // 1) cast x to f16
  cast_f32_f16_kernel<<<blocks(ML * Dc), 256, 0, stream>>>(x, xh, ML * Dc);
  // 2) transpose+cast weights into B-fragment-friendly [N, K] layout
  transpose_w_kernel<<<blocks((size_t)Dc * Hc * HDc), 256, 0, stream>>>(Wq, wqt, Dc, Hc * HDc);
  transpose_w_kernel<<<blocks((size_t)Dc * KVc * HDc), 256, 0, stream>>>(Wk, wkt, Dc, KVc * HDc);
  transpose_w_kernel<<<blocks((size_t)Dc * KVc * HDc), 256, 0, stream>>>(Wv, wvt, Dc, KVc * HDc);
  transpose_w_kernel<<<blocks((size_t)(Hc * HDc) * Dc), 256, 0, stream>>>(Wo, wot, Hc * HDc, Dc);

  // 3) QKV projections (WMMA f16 -> f32); wave tile = 32x64
  auto gemm_blocks = [](int M, int N) { return (unsigned)(((M / 32) * (N / 64) + 7) / 8); };
  wmma_gemm_kernel<<<gemm_blocks((int)ML, Hc * HDc), 256, 0, stream>>>(
      xh, wqt, qf32, (int)ML, Hc * HDc, Dc);
  wmma_gemm_kernel<<<gemm_blocks((int)ML, KVc * HDc), 256, 0, stream>>>(
      xh, wkt, kf32, (int)ML, KVc * HDc, Dc);
  wmma_gemm_kernel<<<gemm_blocks((int)ML, KVc * HDc), 256, 0, stream>>>(
      xh, wvt, vf32, (int)ML, KVc * HDc, Dc);

  // 4) RoPE (Q, K) -> f16 ; 5) V transpose -> [B, KV, HD, LP] f16
  rope_f16_kernel<<<blocks(ML * Hc * (HDc / 2)), 256, 0, stream>>>(qf32, qh, Hc);
  rope_f16_kernel<<<blocks(ML * KVc * (HDc / 2)), 256, 0, stream>>>(kf32, kh, KVc);
  transpose_v_kernel<<<blocks(ML * KVc * HDc), 256, 0, stream>>>(vf32, vt);

  // 6) Flash attention (one wave per (head, 16-query tile))
  {
    unsigned waves = (unsigned)(Bc * Hc * (Lc / 16));
    flash_attn_kernel<<<(waves + 7) / 8, 256, 0, stream>>>(qh, kh, vt, ah);
  }

  // 7) Output projection straight into d_out (f32)
  wmma_gemm_kernel<<<gemm_blocks((int)ML, Dc), 256, 0, stream>>>(
      ah, wot, out, (int)ML, Dc, Hc * HDc);
}
